// ConvolvedAttention_50173807952823
// MI455X (gfx1250) — compile-verified, matrix operand order checked by reading, not run on hardware
//
#include <hip/hip_runtime.h>
#include <hip/hip_bf16.h>
#include <math.h>

// Problem constants (from reference)
#define L_SEQ 8192
#define NB    2
#define EMB   256
#define NH    8
#define HD    32
#define MROWS (L_SEQ * NB)      // 16384 token rows
#define NEGBIG (-1.0e9f)

typedef __attribute__((ext_vector_type(16))) __bf16 bf16x16;
typedef __attribute__((ext_vector_type(8)))  float  f32x8;

__device__ __forceinline__ f32x8 wmma_bf16(bf16x16 a, bf16x16 b, f32x8 c) {
  // D = A(16x32 bf16) * B(32x16 bf16) + C(16x16 f32)
  return __builtin_amdgcn_wmma_f32_16x16x32_bf16(
      /*neg_a=*/false, a, /*neg_b=*/false, b,
      /*c_mod=*/(short)0, c, /*reuse_a=*/false, /*reuse_b=*/false);
}

// ---- CDNA5 async global->LDS staging (ASYNCcnt path, ISA ch.10/15.18) ----
// VDST operand = LDS byte offset (low 32 bits of the flat shared pointer).
__device__ __forceinline__ void async_load_lds_b128(unsigned lds_off,
                                                    const void* gaddr) {
  unsigned long long a = (unsigned long long)(uintptr_t)gaddr;
  asm volatile("global_load_async_to_lds_b128 %0, %1, off"
               :: "v"(lds_off), "v"(a) : "memory");
}
__device__ __forceinline__ void async_load_lds_b64(unsigned lds_off,
                                                   const void* gaddr) {
  unsigned long long a = (unsigned long long)(uintptr_t)gaddr;
  asm volatile("global_load_async_to_lds_b64 %0, %1, off"
               :: "v"(lds_off), "v"(a) : "memory");
}
__device__ __forceinline__ void wait_asynccnt0() {
  asm volatile("s_wait_asynccnt 0" ::: "memory");
}

// CDNA5 16-bit A/B fragment (ISA 7.12.2): per lane, K-chunks
// [8*half, 8*half+8) and [16+8*half, 16+8*half+8) from a contiguous
// 32-element row (row index = lane%16 for A's M / B's N).
__device__ __forceinline__ bf16x16 frag_row_bf16(const __bf16* row, int half) {
  bf16x16 a;
#pragma unroll
  for (int j = 0; j < 8; ++j) {
    a[j]     = row[8 * half + j];
    a[8 + j] = row[16 + 8 * half + j];
  }
  return a;
}

__device__ __forceinline__ bf16x16 frag_row_f32(const float* row, int half) {
  bf16x16 a;
#pragma unroll
  for (int j = 0; j < 8; ++j) {
    a[j]     = (__bf16)row[8 * half + j];
    a[8 + j] = (__bf16)row[16 + 8 * half + j];
  }
  return a;
}

// Row-wise reductions across the 16 lanes of one C-fragment half.
// wave32: xor masks < 16 never cross the half boundary.
__device__ __forceinline__ float redmax16(float v) {
#pragma unroll
  for (int m = 1; m < 16; m <<= 1) v = fmaxf(v, __shfl_xor(v, m, 32));
  return v;
}
__device__ __forceinline__ float redsum16(float v) {
#pragma unroll
  for (int m = 1; m < 16; m <<= 1) v += __shfl_xor(v, m, 32);
  return v;
}

// ---------------------------------------------------------------------------
// Kernel 1: QKV projection. grid = (1024 m-tiles, 3 matrices), block = 128.
// Y = X(16384x256) * W^T + b, emitted as bf16 in [n][h][l][d] layout.
// q additionally scaled by 1/sqrt(hd).
// ---------------------------------------------------------------------------
__global__ __launch_bounds__(128) void qkv_proj_kernel(
    const float* __restrict__ qin, const float* __restrict__ kin,
    const float* __restrict__ vin, const float* __restrict__ w,
    const float* __restrict__ bias, __bf16* __restrict__ qb,
    __bf16* __restrict__ kb, __bf16* __restrict__ vb) {
  __shared__ __attribute__((aligned(16))) float Xs[16 * 32];

  const int mat  = blockIdx.y;             // 0=q, 1=k, 2=v
  const int mt   = blockIdx.x;             // 16-row tile of the 16384 rows
  const int tid  = threadIdx.x;
  const int wave = tid >> 5;
  const int lane = tid & 31;
  const int half = lane >> 4;
  const int ln   = lane & 15;

  const float* X = (mat == 0) ? qin : (mat == 1) ? kin : vin;
  const float* W = w + (size_t)mat * EMB * EMB;
  const float* B = bias + mat * EMB;
  __bf16* dst    = (mat == 0) ? qb : (mat == 1) ? kb : vb;
  const float scale = (mat == 0) ? 0.17677669529663687f : 1.0f;  // 1/sqrt(32)

  // Per-thread staging slot: thread t moves 4 floats (one b128).
  const int sr = tid >> 3;          // tile row 0..15
  const int sc = (tid & 7) * 4;     // tile col 0..28
  const unsigned lds_dst = (unsigned)(uintptr_t)&Xs[sr * 32 + sc];
  const float* src_row = X + (size_t)(mt * 16 + sr) * EMB + sc;

  f32x8 acc[4];
  f32x8 z = {};
#pragma unroll
  for (int ft = 0; ft < 4; ++ft) acc[ft] = z;

  for (int ks = 0; ks < 8; ++ks) {  // K = 256 in steps of 32
    __syncthreads();
    // async global -> LDS stage of the X tile (16 x 32 fp32), no VGPR round trip
    async_load_lds_b128(lds_dst, src_row + ks * 32);
    wait_asynccnt0();
    __syncthreads();

    bf16x16 a = frag_row_f32(&Xs[ln * 32], half);  // A from LDS (convert to bf16)
#pragma unroll
    for (int ft = 0; ft < 4; ++ft) {
      int f0 = (wave * 4 + ft) * 16;
      const float* wrow = &W[(size_t)(f0 + ln) * EMB + ks * 32];
      if (ks < 7) __builtin_prefetch(wrow + 32, 0, 0);  // global_prefetch_b8
      bf16x16 bw = frag_row_f32(wrow, half);            // B[k][n] = W[f0+n][k]
      acc[ft] = wmma_bf16(a, bw, acc[ft]);
    }
  }

  // Write out in [n][h][l][d] bf16 layout, bias added (and q scaled).
#pragma unroll
  for (int ft = 0; ft < 4; ++ft) {
    int f  = (wave * 4 + ft) * 16 + ln;
    float bv = B[f];
    int hh = f >> 5, d = f & 31;
#pragma unroll
    for (int r = 0; r < 8; ++r) {
      int m = mt * 16 + r + 8 * half;   // C layout: VGPR r -> M = r + 8*half
      int l = m >> 1, nn = m & 1;       // m = l*NB + n (NB == 2)
      dst[(((size_t)(nn * NH + hh)) * L_SEQ + l) * HD + d] =
          (__bf16)((acc[ft][r] + bv) * scale);
    }
  }
}

// ---------------------------------------------------------------------------
// Kernel 2: banded attention. One wave handles one (n, h, 16-query tile).
// Scores: 3 WMMAs over 48 keys; softmax in C-fragment registers; P staged
// through LDS (C-layout -> A-layout); O = P x V via 4 WMMAs (K padded to 64).
// ---------------------------------------------------------------------------
__global__ __launch_bounds__(256) void attn_kernel(
    const __bf16* __restrict__ qb, const __bf16* __restrict__ kb,
    const __bf16* __restrict__ vb, __bf16* __restrict__ att) {
  __shared__ __attribute__((aligned(16))) __bf16 Pb[8][16 * 64];

  const int wave = threadIdx.x >> 5;
  const int lane = threadIdx.x & 31;
  const int half = lane >> 4;
  const int ln   = lane & 15;

  const int g  = blockIdx.x * 8 + wave;  // 8192 waves total
  const int n  = g >> 12;                // / (H*512)
  const int h  = (g >> 9) & 7;
  const int lt = g & 511;
  const int l0 = lt * 16;

  const size_t head = ((size_t)(n * NH + h)) * L_SEQ * HD;
  const __bf16* qh = qb + head;
  const __bf16* kh = kb + head;
  const __bf16* vh = vb + head;

  // A = scaled Q tile (16 x 32), contiguous per query row.
  bf16x16 aq = frag_row_bf16(qh + (size_t)(l0 + ln) * HD, half);

  // Scores: S[:, 16t + n'] = q . k[l0 - 32 + 16t + n']   (B[k=d][n'=key])
  f32x8 z = {};
  int key0 = l0 - 32 + ln; if (key0 < 0) key0 = 0;
  int key1 = l0 - 16 + ln; if (key1 < 0) key1 = 0;
  int key2 = l0 + ln;
  f32x8 s0 = wmma_bf16(aq, frag_row_bf16(kh + (size_t)key0 * HD, half), z);
  f32x8 s1 = wmma_bf16(aq, frag_row_bf16(kh + (size_t)key1 * HD, half), z);
  f32x8 s2 = wmma_bf16(aq, frag_row_bf16(kh + (size_t)key2 * HD, half), z);

  // Softmax over the 33 valid window positions per query row.
#pragma unroll
  for (int r = 0; r < 8; ++r) {
    const int i = r + 8 * half;     // query row within tile, l = l0 + i
    float sv0 = s0[r], sv1 = s1[r], sv2 = s2[r];
    // col c = 16t + ln; key = l0 - 32 + c; valid iff i <= c <= i+32 && key >= 0
    bool v0 = (ln >= i)      && (l0 - 32 + ln >= 0);
    bool v1 = (16 + ln >= i) && (l0 - 16 + ln >= 0);
    bool v2 = (ln <= i);
    sv0 = v0 ? sv0 : NEGBIG;
    sv1 = v1 ? sv1 : NEGBIG;
    sv2 = v2 ? sv2 : NEGBIG;
    float mx = redmax16(fmaxf(fmaxf(sv0, sv1), sv2));
    float e0 = __expf(sv0 - mx);
    float e1 = __expf(sv1 - mx);
    float e2 = __expf(sv2 - mx);
    float inv = 1.0f / redsum16(e0 + e1 + e2);
    __bf16* prow = &Pb[wave][i * 64];
    prow[ln]      = (__bf16)(e0 * inv);
    prow[16 + ln] = (__bf16)(e1 * inv);
    prow[32 + ln] = (__bf16)(e2 * inv);
    prow[48 + ln] = (__bf16)0.0f;     // zero-pad K 48..63
  }

  // O = P(16x64) x V(64x32): 2 K-tiles x 2 N-tiles of WMMA.
  f32x8 accO[2];
  accO[0] = z; accO[1] = z;
#pragma unroll
  for (int kt = 0; kt < 2; ++kt) {
    bf16x16 ap = frag_row_bf16(&Pb[wave][ln * 64 + kt * 32], half);
#pragma unroll
    for (int nt = 0; nt < 2; ++nt) {
      const int d = nt * 16 + ln;
      bf16x16 bv;
#pragma unroll
      for (int j = 0; j < 8; ++j) {
        int k0 = 8 * half + j;
        int k1 = 16 + 8 * half + j;
        int kk0 = l0 - 32 + kt * 32 + k0;
        int kk1 = l0 - 32 + kt * 32 + k1;
        kk0 = (kk0 < 0) ? 0 : (kk0 > L_SEQ - 1 ? L_SEQ - 1 : kk0);
        kk1 = (kk1 < 0) ? 0 : (kk1 > L_SEQ - 1 ? L_SEQ - 1 : kk1);
        bv[j]     = vh[(size_t)kk0 * HD + d];   // B[k=key][n'=d]
        bv[8 + j] = vh[(size_t)kk1 * HD + d];
      }
      accO[nt] = wmma_bf16(ap, bv, accO[nt]);
    }
  }

  // att row layout [m = l*NB + n][f = h*32 + d], bf16
#pragma unroll
  for (int nt = 0; nt < 2; ++nt) {
#pragma unroll
    for (int r = 0; r < 8; ++r) {
      int i = r + 8 * half;
      int l = l0 + i;
      int d = nt * 16 + ln;
      att[((size_t)l * NB + n) * EMB + h * HD + d] = (__bf16)accO[nt][r];
    }
  }
}

// ---------------------------------------------------------------------------
// Kernel 3: output projection. Y(fp32) = att(bf16,16384x256) * Wo^T + bo.
// grid = 1024 m-tiles, block = 128 (4 waves x 4 f-tiles each).
// ---------------------------------------------------------------------------
__global__ __launch_bounds__(128) void out_proj_kernel(
    const __bf16* __restrict__ att, const float* __restrict__ w,
    const float* __restrict__ bias, float* __restrict__ out) {
  __shared__ __attribute__((aligned(16))) __bf16 Xs[16 * 32];

  const int mt   = blockIdx.x;
  const int tid  = threadIdx.x;
  const int wave = tid >> 5;
  const int lane = tid & 31;
  const int half = lane >> 4;
  const int ln   = lane & 15;

  // Per-thread staging slot: thread t moves 4 bf16 (one b64).
  const int sr = tid >> 3;
  const int sc = (tid & 7) * 4;
  const unsigned lds_dst = (unsigned)(uintptr_t)&Xs[sr * 32 + sc];
  const __bf16* src_row = att + (size_t)(mt * 16 + sr) * EMB + sc;

  f32x8 acc[4];
  f32x8 z = {};
#pragma unroll
  for (int ft = 0; ft < 4; ++ft) acc[ft] = z;

  for (int ks = 0; ks < 8; ++ks) {
    __syncthreads();
    async_load_lds_b64(lds_dst, src_row + ks * 32);
    wait_asynccnt0();
    __syncthreads();

    bf16x16 a = frag_row_bf16(&Xs[ln * 32], half);
#pragma unroll
    for (int ft = 0; ft < 4; ++ft) {
      int f0 = (wave * 4 + ft) * 16;
      bf16x16 bw = frag_row_f32(&w[(size_t)(f0 + ln) * EMB + ks * 32], half);
      acc[ft] = wmma_bf16(a, bw, acc[ft]);
    }
  }

#pragma unroll
  for (int ft = 0; ft < 4; ++ft) {
    int f = (wave * 4 + ft) * 16 + ln;
    float bv = bias[f];
#pragma unroll
    for (int r = 0; r < 8; ++r) {
      int m = mt * 16 + r + 8 * half;
      out[(size_t)m * EMB + f] = acc[ft][r] + bv;
    }
  }
}

// ---------------------------------------------------------------------------
extern "C" void kernel_launch(void* const* d_in, const int* in_sizes, int n_in,
                              void* d_out, int out_size, void* d_ws,
                              size_t ws_size, hipStream_t stream) {
  (void)in_sizes; (void)n_in; (void)out_size; (void)ws_size;

  const float* query      = (const float*)d_in[0];
  const float* key        = (const float*)d_in[1];
  const float* value      = (const float*)d_in[2];
  const float* in_proj_w  = (const float*)d_in[3];
  const float* in_proj_b  = (const float*)d_in[4];
  const float* out_proj_w = (const float*)d_in[5];
  const float* out_proj_b = (const float*)d_in[6];
  float* out = (float*)d_out;

  // Workspace: qb | kb | vb | att, each 4 Mi bf16 elements (8 MiB) -> 32 MiB.
  __bf16* ws  = (__bf16*)d_ws;
  const size_t per = (size_t)NB * NH * L_SEQ * HD;
  __bf16* qb  = ws;
  __bf16* kb  = ws + per;
  __bf16* vb  = ws + 2 * per;
  __bf16* att = ws + 3 * per;

  qkv_proj_kernel<<<dim3(MROWS / 16, 3), 128, 0, stream>>>(
      query, key, value, in_proj_w, in_proj_b, qb, kb, vb);

  attn_kernel<<<(NB * NH * (L_SEQ / 16)) / 8, 256, 0, stream>>>(qb, kb, vb, att);

  out_proj_kernel<<<MROWS / 16, 128, 0, stream>>>(att, out_proj_w, out_proj_b, out);
}
